// ECMTSpikingAttention_65609920413870
// MI455X (gfx1250) — compile-verified, matrix operand order checked by reading, not run on hardware
//
#include <hip/hip_runtime.h>

typedef __attribute__((ext_vector_type(16))) _Float16 v16h;
typedef __attribute__((ext_vector_type(8)))  _Float16 v8h;
typedef __attribute__((ext_vector_type(8)))  float    v8f;

#define B_   4
#define S_   1024
#define D_   1024
#define H_   16
#define HD_  64
#define M_   (B_ * S_)          // 4096 rows
#define WAVES 4                 // waves per block
#define TB   (WAVES * 32)       // 128 threads
#define NT   4                  // n-tiles per wave (register blocking)

// ---------------------------------------------------------------------------
// Fragment load: 16-bit A/B matrix layout (ISA 7.12.2).
// lane = (idx15 = lane&15, half = lane>>4); element e of the v16h maps to
//   k = k0 + half*8 + (e < 8 ? e : e + 8)
// i.e. two contiguous 8-half (16B) runs at +0 and +16 halves.
// ---------------------------------------------------------------------------
__device__ __forceinline__ v16h ld_frag16(const _Float16* __restrict__ p) {
  v8h a = *(const v8h*)(p);
  v8h b = *(const v8h*)(p + 16);
  v16h f;
#pragma unroll
  for (int i = 0; i < 8; ++i) { f[i] = a[i]; f[i + 8] = b[i]; }
  return f;
}

// Multi-threshold neuron: mem = x/2; 4 thresholds 1..4; spikes += nw[i]/4.
__device__ __forceinline__ float mtn4(float x, float n0, float n1, float n2, float n3) {
  float mem = x * 0.5f, sp = 0.0f, fired;
  fired = (mem >= 1.0f) ? 1.0f : 0.0f; sp += fired * n0 * 0.25f; mem -= fired * 1.0f;
  fired = (mem >= 2.0f) ? 1.0f : 0.0f; sp += fired * n1 * 0.25f; mem -= fired * 2.0f;
  fired = (mem >= 3.0f) ? 1.0f : 0.0f; sp += fired * n2 * 0.25f; mem -= fired * 3.0f;
  fired = (mem >= 4.0f) ? 1.0f : 0.0f; sp += fired * n3 * 0.25f; mem -= fired * 4.0f;
  return sp;
}

// ---------------------------------------------------------------------------
// fp32 -> fp16 conversion
// ---------------------------------------------------------------------------
__global__ void cvt_f32_f16(const float* __restrict__ in, _Float16* __restrict__ out, int n) {
  int i = blockIdx.x * blockDim.x + threadIdx.x;
  if (i < n) out[i] = (_Float16)in[i];
}

// ---------------------------------------------------------------------------
// C[m,n] = MTN( sum_k A[m,k] * W[n,k] + bias[n] )
// One wave computes a 16x64 strip: 1 A-tile reused across NT=4 B-tiles.
// ---------------------------------------------------------------------------
template <typename OutT>
__global__ void gemm_bias_mtn(const _Float16* __restrict__ A,
                              const _Float16* __restrict__ W,
                              const float* __restrict__ bias,
                              const float* __restrict__ nw,
                              OutT* __restrict__ out, int Ngroups) {
  const int wave = threadIdx.x >> 5;
  const int lane = threadIdx.x & 31;
  const int tile = blockIdx.x * WAVES + wave;
  const int tm = tile / Ngroups, tg = tile % Ngroups;   // tg = group of 4 n-tiles
  const int idx15 = lane & 15, half = lane >> 4;

  const _Float16* arow = A + (size_t)(tm * 16 + idx15) * D_;
  const _Float16* brow[NT];
#pragma unroll
  for (int nt = 0; nt < NT; ++nt)
    brow[nt] = W + (size_t)(tg * (16 * NT) + nt * 16 + idx15) * D_;

  v8f acc[NT] = {};
#pragma unroll 2
  for (int k0 = 0; k0 < D_; k0 += 32) {
    __builtin_prefetch(arow + k0 + 256, 0, 0);   // global_prefetch_b8
    v16h af = ld_frag16(arow + k0 + half * 8);
#pragma unroll
    for (int nt = 0; nt < NT; ++nt) {
      v16h bf = ld_frag16(brow[nt] + k0 + half * 8);
      acc[nt] = __builtin_amdgcn_wmma_f32_16x16x32_f16(false, af, false, bf,
                                                       (short)0, acc[nt], false, false);
    }
  }

  const int mbase = tm * 16 + half * 8;      // C: M = vgpr_j + 8*(lane/16)
  const float n0 = nw[0], n1 = nw[1], n2 = nw[2], n3 = nw[3];
#pragma unroll
  for (int nt = 0; nt < NT; ++nt) {
    const int n = tg * (16 * NT) + nt * 16 + idx15;   // C: N = lane%16
    const float bv = bias[n];
#pragma unroll
    for (int j = 0; j < 8; ++j) {
      float v = mtn4(acc[nt][j] + bv, n0, n1, n2, n3);
      out[(size_t)(mbase + j) * D_ + n] = (OutT)v;
    }
  }
}

// ---------------------------------------------------------------------------
// ECM compensation: y = s - (s @ Wc^T) + bc, f16 in / f16 out, 16x64 strip.
// ---------------------------------------------------------------------------
__global__ void gemm_comp(const _Float16* __restrict__ Sm,
                          const _Float16* __restrict__ Wc,
                          const float* __restrict__ bc,
                          _Float16* __restrict__ out, int Ngroups) {
  const int wave = threadIdx.x >> 5;
  const int lane = threadIdx.x & 31;
  const int tile = blockIdx.x * WAVES + wave;
  const int tm = tile / Ngroups, tg = tile % Ngroups;
  const int idx15 = lane & 15, half = lane >> 4;

  const _Float16* arow = Sm + (size_t)(tm * 16 + idx15) * D_;
  const _Float16* brow[NT];
#pragma unroll
  for (int nt = 0; nt < NT; ++nt)
    brow[nt] = Wc + (size_t)(tg * (16 * NT) + nt * 16 + idx15) * D_;

  v8f acc[NT] = {};
#pragma unroll 2
  for (int k0 = 0; k0 < D_; k0 += 32) {
    v16h af = ld_frag16(arow + k0 + half * 8);
#pragma unroll
    for (int nt = 0; nt < NT; ++nt) {
      v16h bf = ld_frag16(brow[nt] + k0 + half * 8);
      acc[nt] = __builtin_amdgcn_wmma_f32_16x16x32_f16(false, af, false, bf,
                                                       (short)0, acc[nt], false, false);
    }
  }

  const int mbase = tm * 16 + half * 8;
#pragma unroll
  for (int nt = 0; nt < NT; ++nt) {
    const int n = tg * (16 * NT) + nt * 16 + idx15;
    const float bv = bc[n];
#pragma unroll
    for (int j = 0; j < 8; ++j) {
      size_t idx = (size_t)(mbase + j) * D_ + n;
      float sv = (float)Sm[idx];
      out[idx] = (_Float16)(sv - acc[nt][j] + bv);
    }
  }
}

// ---------------------------------------------------------------------------
// Attention logits: attn[b,h,q,k] = (q . k) * hd^-0.5 / (temp + 1e-8)
// One wave: 1 Q-tile x 4 K-tiles (16x64 of the score matrix). K-dim = 64.
// ---------------------------------------------------------------------------
__global__ void attn_scores(const _Float16* __restrict__ qf,
                            const _Float16* __restrict__ kf,
                            const float* __restrict__ temperature,
                            float* __restrict__ attn) {
  const int wave = threadIdx.x >> 5;
  const int lane = threadIdx.x & 31;
  int id = blockIdx.x * WAVES + wave;
  const int tk4 = id & 15; id >>= 4;    // group of 4 key tiles (64 cols)
  const int tq  = id & 63; id >>= 6;
  const int h   = id & 15; const int b = id >> 4;
  const int idx15 = lane & 15, half = lane >> 4;

  const _Float16* qrow = qf + ((size_t)(b * S_) + tq * 16 + idx15) * D_ + h * HD_;
  const _Float16* krow[NT];
#pragma unroll
  for (int nt = 0; nt < NT; ++nt)
    krow[nt] = kf + ((size_t)(b * S_) + tk4 * 64 + nt * 16 + idx15) * D_ + h * HD_;

  v8f acc[NT] = {};
#pragma unroll
  for (int k0 = 0; k0 < HD_; k0 += 32) {
    v16h af = ld_frag16(qrow + k0 + half * 8);
#pragma unroll
    for (int nt = 0; nt < NT; ++nt) {
      v16h bf = ld_frag16(krow[nt] + k0 + half * 8);
      acc[nt] = __builtin_amdgcn_wmma_f32_16x16x32_f16(false, af, false, bf,
                                                       (short)0, acc[nt], false, false);
    }
  }

  const float sc = 0.125f / (temperature[0] + 1e-8f);  // hd^-0.5 = 0.125
  const int mq = tq * 16 + half * 8;
  float* base = attn + ((size_t)(b * H_ + h)) * S_ * S_;
#pragma unroll
  for (int nt = 0; nt < NT; ++nt) {
    const int n = tk4 * 64 + nt * 16 + idx15;
#pragma unroll
    for (int j = 0; j < 8; ++j)
      base[(size_t)(mq + j) * S_ + n] = acc[nt][j] * sc;
  }
}

// ---------------------------------------------------------------------------
// In-place row softmax over rows of length S_. One 256-thread block per row.
// ---------------------------------------------------------------------------
__global__ void softmax_rows(float* __restrict__ attn) {
  __shared__ float red[256];
  float* p = attn + (size_t)blockIdx.x * S_;
  const int t = threadIdx.x;

  float m = -1e30f;
  for (int i = t; i < S_; i += 256) m = fmaxf(m, p[i]);
  red[t] = m; __syncthreads();
  for (int s2 = 128; s2 > 0; s2 >>= 1) {
    if (t < s2) red[t] = fmaxf(red[t], red[t + s2]);
    __syncthreads();
  }
  m = red[0]; __syncthreads();

  float s = 0.0f;
  for (int i = t; i < S_; i += 256) { float e = __expf(p[i] - m); p[i] = e; s += e; }
  red[t] = s; __syncthreads();
  for (int s2 = 128; s2 > 0; s2 >>= 1) {
    if (t < s2) red[t] += red[t + s2];
    __syncthreads();
  }
  const float inv = 1.0f / red[0];
  for (int i = t; i < S_; i += 256) p[i] *= inv;
}

// ---------------------------------------------------------------------------
// Per-batch transpose: vT[b][d][s] = vf[b][s][d]  (16x16 LDS tiles, f16).
// Makes attn_pv B-operand fragments contiguous b128 loads.
// ---------------------------------------------------------------------------
__global__ void transpose_v(const _Float16* __restrict__ vf, _Float16* __restrict__ vT) {
  __shared__ _Float16 tile[16][17];
  int id = blockIdx.x;
  const int td = id & 63; id >>= 6;   // d tile
  const int ts = id & 63; id >>= 6;   // s tile
  const int b = id;
  const int tx = threadIdx.x & 15, ty = threadIdx.x >> 4;
  tile[ty][tx] = vf[((size_t)(b * S_) + ts * 16 + ty) * D_ + td * 16 + tx];
  __syncthreads();
  vT[((size_t)(b * D_) + td * 16 + ty) * S_ + ts * 16 + tx] = tile[tx][ty];
}

// ---------------------------------------------------------------------------
// attn_out[b,q,h*64+d] = sum_k P[b,h,q,k] * vT[b, h*64+d, k]   (K = 1024)
// One wave: 1 P-tile reused across all 4 d-tiles of the head.
// ---------------------------------------------------------------------------
__global__ void attn_pv(const float* __restrict__ attn,
                        const _Float16* __restrict__ vT,
                        _Float16* __restrict__ aout) {
  const int wave = threadIdx.x >> 5;
  const int lane = threadIdx.x & 31;
  int id = blockIdx.x * WAVES + wave;
  const int tq = id & 63; id >>= 6;
  const int h  = id & 15; const int b = id >> 4;
  const int idx15 = lane & 15, half = lane >> 4;

  const float* prow = attn + ((size_t)(b * H_ + h) * S_ + tq * 16 + idx15) * S_;
  const _Float16* brow[NT];
#pragma unroll
  for (int nt = 0; nt < NT; ++nt)
    brow[nt] = vT + ((size_t)(b * D_) + h * HD_ + nt * 16 + idx15) * S_;

  v8f acc[NT] = {};
  for (int k0 = 0; k0 < S_; k0 += 32) {
    const int kb = k0 + half * 8;
    v16h af;
    const float* pa = prow + kb;
#pragma unroll
    for (int i = 0; i < 8; ++i) {
      af[i]     = (_Float16)pa[i];
      af[i + 8] = (_Float16)pa[16 + i];
    }
#pragma unroll
    for (int nt = 0; nt < NT; ++nt) {
      v16h bf = ld_frag16(brow[nt] + kb);
      acc[nt] = __builtin_amdgcn_wmma_f32_16x16x32_f16(false, af, false, bf,
                                                       (short)0, acc[nt], false, false);
    }
  }

  const int mq = tq * 16 + half * 8;
#pragma unroll
  for (int nt = 0; nt < NT; ++nt) {
    const int n = h * HD_ + nt * 16 + idx15;
#pragma unroll
    for (int j = 0; j < 8; ++j)
      aout[((size_t)(b * S_) + mq + j) * D_ + n] = (_Float16)acc[nt][j];
  }
}

// ---------------------------------------------------------------------------
// Launch
// ---------------------------------------------------------------------------
extern "C" void kernel_launch(void* const* d_in, const int* in_sizes, int n_in,
                              void* d_out, int out_size, void* d_ws, size_t ws_size,
                              hipStream_t stream) {
  const float* query = (const float*)d_in[0];
  const float* key   = (const float*)d_in[1];
  const float* value = (const float*)d_in[2];
  const float* Wq  = (const float*)d_in[3];   const float* bq  = (const float*)d_in[4];
  const float* Wk  = (const float*)d_in[5];   const float* bk  = (const float*)d_in[6];
  const float* Wv  = (const float*)d_in[7];   const float* bv  = (const float*)d_in[8];
  const float* Wo  = (const float*)d_in[9];   const float* bo  = (const float*)d_in[10];
  const float* Wqc = (const float*)d_in[11];  const float* bqc = (const float*)d_in[12];
  const float* Wkc = (const float*)d_in[13];  const float* bkc = (const float*)d_in[14];
  const float* Wvc = (const float*)d_in[15];  const float* bvc = (const float*)d_in[16];
  const float* nw_q = (const float*)d_in[17];
  const float* nw_k = (const float*)d_in[18];
  const float* nw_v = (const float*)d_in[19];
  const float* nw_o = (const float*)d_in[20];
  const float* temperature = (const float*)d_in[21];

  char* ws = (char*)d_ws;
  const size_t SZ_ACT = (size_t)M_ * D_ * sizeof(_Float16);  // 8 MB
  const size_t SZ_W   = (size_t)D_ * D_ * sizeof(_Float16);  // 2 MB
  size_t off = 0;
  _Float16* hQ   = (_Float16*)(ws + off); off += SZ_ACT;
  _Float16* hK   = (_Float16*)(ws + off); off += SZ_ACT;
  _Float16* hV   = (_Float16*)(ws + off); off += SZ_ACT;
  _Float16* hWq  = (_Float16*)(ws + off); off += SZ_W;
  _Float16* hWk  = (_Float16*)(ws + off); off += SZ_W;
  _Float16* hWv  = (_Float16*)(ws + off); off += SZ_W;
  _Float16* hWo  = (_Float16*)(ws + off); off += SZ_W;
  _Float16* hWqc = (_Float16*)(ws + off); off += SZ_W;
  _Float16* hWkc = (_Float16*)(ws + off); off += SZ_W;
  _Float16* hWvc = (_Float16*)(ws + off); off += SZ_W;
  _Float16* sQ   = (_Float16*)(ws + off); off += SZ_ACT;
  _Float16* sK   = (_Float16*)(ws + off); off += SZ_ACT;
  _Float16* sV   = (_Float16*)(ws + off); off += SZ_ACT;
  _Float16* qf   = (_Float16*)(ws + off); off += SZ_ACT;
  _Float16* kf   = (_Float16*)(ws + off); off += SZ_ACT;
  _Float16* vf   = (_Float16*)(ws + off); off += SZ_ACT;
  _Float16* vT   = (_Float16*)(ws + off); off += SZ_ACT;
  _Float16* aout = (_Float16*)(ws + off); off += SZ_ACT;

  float* out0 = (float*)d_out;                      // [B,S,D] fp32
  float* attn = out0 + (size_t)M_ * D_;             // [B,H,S,S] fp32

  const int nAct = M_ * D_;   // 4.19M
  const int nW   = D_ * D_;   // 1.05M
  cvt_f32_f16<<<(nAct + 255) / 256, 256, 0, stream>>>(query, hQ, nAct);
  cvt_f32_f16<<<(nAct + 255) / 256, 256, 0, stream>>>(key,   hK, nAct);
  cvt_f32_f16<<<(nAct + 255) / 256, 256, 0, stream>>>(value, hV, nAct);
  cvt_f32_f16<<<(nW + 255) / 256, 256, 0, stream>>>(Wq,  hWq,  nW);
  cvt_f32_f16<<<(nW + 255) / 256, 256, 0, stream>>>(Wk,  hWk,  nW);
  cvt_f32_f16<<<(nW + 255) / 256, 256, 0, stream>>>(Wv,  hWv,  nW);
  cvt_f32_f16<<<(nW + 255) / 256, 256, 0, stream>>>(Wo,  hWo,  nW);
  cvt_f32_f16<<<(nW + 255) / 256, 256, 0, stream>>>(Wqc, hWqc, nW);
  cvt_f32_f16<<<(nW + 255) / 256, 256, 0, stream>>>(Wkc, hWkc, nW);
  cvt_f32_f16<<<(nW + 255) / 256, 256, 0, stream>>>(Wvc, hWvc, nW);

  const int Ngroups = D_ / (16 * NT);                // 16 groups of 64 cols
  const int gemmWaves = (M_ / 16) * Ngroups;         // 4096
  const int gemmBlocks = gemmWaves / WAVES;          // 1024

  // Spiking projections: s = MTN(X W^T + b)
  gemm_bias_mtn<_Float16><<<gemmBlocks, TB, 0, stream>>>(hQ, hWq, bq, nw_q, sQ, Ngroups);
  gemm_bias_mtn<_Float16><<<gemmBlocks, TB, 0, stream>>>(hK, hWk, bk, nw_k, sK, Ngroups);
  gemm_bias_mtn<_Float16><<<gemmBlocks, TB, 0, stream>>>(hV, hWv, bv, nw_v, sV, Ngroups);

  // ECM compensation: y = s - s Wc^T + bc
  gemm_comp<<<gemmBlocks, TB, 0, stream>>>(sQ, hWqc, bqc, qf, Ngroups);
  gemm_comp<<<gemmBlocks, TB, 0, stream>>>(sK, hWkc, bkc, kf, Ngroups);
  gemm_comp<<<gemmBlocks, TB, 0, stream>>>(sV, hWvc, bvc, vf, Ngroups);

  // Attention logits -> d_out attn region
  const int scoreWaves = B_ * H_ * 64 * 16;          // 65536
  attn_scores<<<scoreWaves / WAVES, TB, 0, stream>>>(qf, kf, temperature, attn);

  // Row softmax in place
  softmax_rows<<<B_ * H_ * S_, 256, 0, stream>>>(attn);

  // vT[b][d][s] = vf[b][s][d]
  transpose_v<<<B_ * 64 * 64, 256, 0, stream>>>(vf, vT);

  // attn_out = P @ V
  const int pvWaves = B_ * H_ * 64;                  // 4096
  attn_pv<<<pvWaves / WAVES, TB, 0, stream>>>(attn, vT, aout);

  // output = MTN(attn_out Wo^T + bo) -> d_out region 0 (fp32)
  gemm_bias_mtn<float><<<gemmBlocks, TB, 0, stream>>>(aout, hWo, bo, nw_o, out0, Ngroups);

  (void)in_sizes; (void)n_in; (void)out_size; (void)ws_size;
}
